// AttentionPooling_23691039605148
// MI455X (gfx1250) — matrix operand order repaired and checked by reading, not compiled, not run on hardware
//
#include <hip/hip_runtime.h>
#include <hip/hip_bf16.h>
#include <math.h>

// ---------------------------------------------------------------------------
// AttentionPooling on MI455X (gfx1250, wave32).
//   B=32, N=4096, D=512, H=64.
//   Memory-bound: Z = 256MB streamed twice (~22us @ 23.3 TB/s).
//   GEMM phase uses v_wmma_f32_16x16x32_bf16; f32->bf16 packing is one
//   v_add + v_perm_b32 per pair so VALU stays off the bandwidth-critical path.
// ---------------------------------------------------------------------------

typedef __attribute__((ext_vector_type(16))) __bf16 v16bf;
typedef __attribute__((ext_vector_type(8)))  float  v8f;

union BfPack { unsigned u[8]; v16bf v; };

// f32 -> bf16 (round-half-up) packed pair in one v_perm_b32:
// take high 16 bits of (bits + 0x8000) of each float.
// v_perm_b32 src bytes: S1 = bytes 0..3, S0 = bytes 4..7.
// sel 0x07060302 -> dst = { hi[31:16], lo[31:16] }.
__device__ __forceinline__ unsigned pk_bf16(float lo, float hi) {
    unsigned a = __float_as_uint(lo) + 0x8000u;
    unsigned b = __float_as_uint(hi) + 0x8000u;
    return __builtin_amdgcn_perm(b, a, 0x07060302u);
}

// ---------------------------------------------------------------------------
// Kernel 0: swizzle W_proj [512x64] f32 into WMMA B-operand layout (bf16).
// B tile is 32(K) x 16(N). Own k order (not ISA table): lanes 0-15 hold K=0..15, lanes 16-31 K=16..31;
// VGPR v holds K pair {16*half + 2v, +1}. 64 tiles (16 K-steps x 4 N-tiles).
// ws layout: [tile][lane][v] as dwords -> each lane reads 32B contiguous.
// ---------------------------------------------------------------------------
__global__ __launch_bounds__(32) void prep_w(const float* __restrict__ W,
                                             unsigned* __restrict__ wsW) {
    const int lane = threadIdx.x;          // 0..31
    const int blk  = blockIdx.x;           // ks*4 + t, 64 blocks
    const int ks   = blk >> 2;
    const int t    = blk & 3;
    const int col  = t * 16 + (lane & 15); // N index, 0..63
    const int h    = lane >> 4;
    const int k0   = ks * 32 + 16 * h;
#pragma unroll
    for (int v = 0; v < 8; ++v) {
        const int K = k0 + 2 * v;
        wsW[(blk * 32 + lane) * 8 + v] =
            pk_bf16(W[K * 64 + col], W[(K + 1) * 64 + col]);
    }
}

// ---------------------------------------------------------------------------
// Kernel 1: scores = tanh(Z @ Wp + bp) @ Wa + ba   (fused, h never stored)
// One wave = one 16-row tile of Z. 16 K-steps of 32, 4 N-tiles of 16.
// A k order used here (not ISA table order): lane half h, VGPR v -> K=16h+2v
// (matches swizzled B): two contiguous 8-float chunks per lane.
// ---------------------------------------------------------------------------
__global__ __launch_bounds__(256) void scores_kernel(
    const float* __restrict__ Z,
    const float* __restrict__ b_proj,   // [64]
    const float* __restrict__ W_attn,   // [64]
    const float* __restrict__ b_attn,   // [1]
    const unsigned* __restrict__ wsW,   // swizzled W_proj
    float* __restrict__ scores)         // [B*N] (alphas region of d_out)
{
    const int lane = threadIdx.x & 31;
    const int wave = threadIdx.x >> 5;
    const long rowBase = (long)blockIdx.x * 128 + wave * 16; // 16 rows / wave
    const int r16 = lane & 15;
    const int h   = lane >> 4;
    const float* zrow = Z + (rowBase + r16) * 512;

    v8f c[4];
#pragma unroll
    for (int t = 0; t < 4; ++t)
#pragma unroll
        for (int i = 0; i < 8; ++i) c[t][i] = 0.0f;

#pragma unroll 4
    for (int ks = 0; ks < 16; ++ks) {
        const int k0 = ks * 32;
        // A operand: 16 f32 -> 16 bf16 (two contiguous 32B chunks per lane)
        const float4* pA0 = (const float4*)(zrow + k0 + 16 * h);
        const float4* pA1 = (const float4*)(zrow + k0 + 16 * h + 8);
        float4 f0 = pA0[0], f1 = pA0[1];
        float4 f2 = pA1[0], f3 = pA1[1];
        __builtin_prefetch(zrow + k0 + 32 + 8 * h, 0, 3); // next K-step, near
        BfPack a;
        a.u[0] = pk_bf16(f0.x, f0.y); a.u[1] = pk_bf16(f0.z, f0.w);
        a.u[2] = pk_bf16(f1.x, f1.y); a.u[3] = pk_bf16(f1.z, f1.w);
        a.u[4] = pk_bf16(f2.x, f2.y); a.u[5] = pk_bf16(f2.z, f2.w);
        a.u[6] = pk_bf16(f3.x, f3.y); a.u[7] = pk_bf16(f3.z, f3.w);
#pragma unroll
        for (int t = 0; t < 4; ++t) {
            const uint4* pb =
                (const uint4*)(wsW + (((ks * 4 + t) * 32 + lane) * 8));
            uint4 b0 = pb[0], b1 = pb[1];
            BfPack bb;
            bb.u[0] = b0.x; bb.u[1] = b0.y; bb.u[2] = b0.z; bb.u[3] = b0.w;
            bb.u[4] = b1.x; bb.u[5] = b1.y; bb.u[6] = b1.z; bb.u[7] = b1.w;
            c[t] = __builtin_amdgcn_wmma_f32_16x16x32_bf16(
                false, a.v, false, bb.v, (short)0, c[t], false, false);
        }
    }

    // Epilogue: bias + tanh + dot with W_attn.
    // C layout: VGPR r, lanes 0-15 -> (row r, col lane), lanes 16-31 ->
    // (row 8+r, col lane-16). Per-lane column = t*16 + r16.
    float p[8];
#pragma unroll
    for (int r = 0; r < 8; ++r) p[r] = 0.0f;
#pragma unroll
    for (int t = 0; t < 4; ++t) {
        const int col = t * 16 + r16;
        const float bpc = b_proj[col];
        const float wac = W_attn[col];
#pragma unroll
        for (int r = 0; r < 8; ++r)
            p[r] += tanhf(c[t][r] + bpc) * wac;
    }
    // Reduce across the 16-lane half (columns); rows stay separated by half.
#pragma unroll
    for (int r = 0; r < 8; ++r) {
#pragma unroll
        for (int m = 8; m >= 1; m >>= 1)
            p[r] += __shfl_xor(p[r], m, 16);
    }
    if (r16 == 0) {
        const float ba = b_attn[0];
#pragma unroll
        for (int r = 0; r < 8; ++r)
            scores[rowBase + 8 * h + r] = p[r] + ba;  // row = 8h + r
    }
}

// ---------------------------------------------------------------------------
// Kernel 2: in-place softmax over N=4096 per batch. 256 threads, 16 vals each.
// ---------------------------------------------------------------------------
__global__ __launch_bounds__(256) void softmax_kernel(float* __restrict__ a) {
    __shared__ float red[256];
    const int tid = threadIdx.x;
    float* s = a + (long)blockIdx.x * 4096;

    float loc[16];
    float mx = -INFINITY;
#pragma unroll
    for (int i = 0; i < 16; ++i) {
        loc[i] = s[tid + i * 256];
        mx = fmaxf(mx, loc[i]);
    }
    red[tid] = mx; __syncthreads();
    for (int off = 128; off >= 1; off >>= 1) {
        if (tid < off) red[tid] = fmaxf(red[tid], red[tid + off]);
        __syncthreads();
    }
    mx = red[0]; __syncthreads();

    float sum = 0.0f;
#pragma unroll
    for (int i = 0; i < 16; ++i) {
        loc[i] = expf(loc[i] - mx);
        sum += loc[i];
    }
    red[tid] = sum; __syncthreads();
    for (int off = 128; off >= 1; off >>= 1) {
        if (tid < off) red[tid] += red[tid + off];
        __syncthreads();
    }
    const float inv = 1.0f / red[0];
#pragma unroll
    for (int i = 0; i < 16; ++i) s[tid + i * 256] = loc[i] * inv;
}

// ---------------------------------------------------------------------------
// Kernel 3a: pooled partials. grid = B*16 chunks, 512 threads (one per d).
// Deterministic (no float atomics) so graph replays are bit-identical.
// ---------------------------------------------------------------------------
__global__ __launch_bounds__(512) void pool_partial(
    const float* __restrict__ Z, const float* __restrict__ alphas,
    float* __restrict__ partial) {
    const int b  = blockIdx.x >> 4;
    const int ch = blockIdx.x & 15;     // 256 rows per chunk
    const int d  = threadIdx.x;
    const float* zb = Z + ((long)b * 4096 + ch * 256) * 512;
    const float* al = alphas + b * 4096 + ch * 256;
    float acc = 0.0f;
#pragma unroll 4
    for (int n = 0; n < 256; ++n)
        acc = fmaf(al[n], zb[(long)n * 512 + d], acc);
    partial[(b * 16 + ch) * 512 + d] = acc;
}

__global__ __launch_bounds__(256) void pool_reduce(
    const float* __restrict__ partial, float* __restrict__ pooled) {
    const int idx = blockIdx.x * 256 + threadIdx.x; // 0..16383 = b*512+d
    const int b = idx >> 9;
    const int d = idx & 511;
    float acc = 0.0f;
#pragma unroll
    for (int ch = 0; ch < 16; ++ch)
        acc += partial[(b * 16 + ch) * 512 + d];
    pooled[idx] = acc;
}

// ---------------------------------------------------------------------------
extern "C" void kernel_launch(void* const* d_in, const int* in_sizes, int n_in,
                              void* d_out, int out_size, void* d_ws, size_t ws_size,
                              hipStream_t stream) {
    const float* Z      = (const float*)d_in[0];  // [32,4096,512]
    const float* W_proj = (const float*)d_in[1];  // [512,64]
    const float* b_proj = (const float*)d_in[2];  // [64]
    const float* W_attn = (const float*)d_in[3];  // [64]
    const float* b_attn = (const float*)d_in[4];  // [1]

    float* pooled = (float*)d_out;                // [32,512]
    float* alphas = (float*)d_out + 32 * 512;     // [32,4096]

    unsigned* wsW   = (unsigned*)d_ws;                       // 64KB swizzled W
    float*  partial = (float*)((char*)d_ws + 64 * 1024);     // 1MB partials

    prep_w       <<<64,   32, 0, stream>>>(W_proj, wsW);
    scores_kernel<<<1024, 256, 0, stream>>>(Z, b_proj, W_attn, b_attn, wsW, alphas);
    softmax_kernel<<<32,  256, 0, stream>>>(alphas);
    pool_partial <<<512,  512, 0, stream>>>(Z, alphas, partial);
    pool_reduce  <<<64,   256, 0, stream>>>(partial, pooled);
}
